// MultiGraphRGCN_11510512354050
// MI455X (gfx1250) — matrix operand order repaired and checked by reading, not compiled, not run on hardware
//
#include <hip/hip_runtime.h>
#include <hip/hip_bf16.h>

// ---------------------------------------------------------------------------
// RGCN (basis decomposition) for MI455X / gfx1250.
//
// Per layer:
//   h[r, dst, :] += norm[r,dst] * x[src, :]     (bf16 packed atomics -> L2-resident h)
//   out = ELU( sum_r h[r,n,:] @ W_r  +  x @ root + bias )
// W_r = sum_b comb[r,b]*bases[b]; root treated as relation 8.
// GEMMs run on v_wmma_f32_16x16x32_bf16 (f32 accumulate).
// ---------------------------------------------------------------------------

#define RGCN_R 8
#define RGCN_B 4
#define RGCN_C 64

typedef __bf16  bf16_t;
typedef bf16_t  v8bf  __attribute__((ext_vector_type(8)));
typedef bf16_t  v16bf __attribute__((ext_vector_type(16)));
typedef float   v8f   __attribute__((ext_vector_type(8)));

static __device__ __forceinline__ unsigned short f2bf_bits(float f) {
    unsigned u = __builtin_bit_cast(unsigned, f);
    unsigned r = u + 0x7FFFu + ((u >> 16) & 1u);   // round-to-nearest-even
    return (unsigned short)(r >> 16);
}
static __device__ __forceinline__ bf16_t f2bf(float f) {
    unsigned short h = f2bf_bits(f);
    return __builtin_bit_cast(bf16_t, h);
}
static __device__ __forceinline__ unsigned pack2bf(float lo, float hi) {
    return (unsigned)f2bf_bits(lo) | ((unsigned)f2bf_bits(hi) << 16);
}

// CDNA5 packed-bf16 global atomic add (no return -> STOREcnt tracked)
static __device__ __forceinline__ void atomic_pk_add_bf16(bf16_t* addr, unsigned data) {
    asm volatile("global_atomic_pk_add_bf16 %0, %1, off"
                 :: "v"(addr), "v"(data) : "memory");
}

// ---------------------------------------------------------------------------
// Utility: zero a float-addressable buffer (grid-stride)
// ---------------------------------------------------------------------------
__global__ void rgcn_zero_kernel(float* __restrict__ p, long long n) {
    long long i = (long long)blockIdx.x * blockDim.x + threadIdx.x;
    long long stride = (long long)gridDim.x * blockDim.x;
    for (; i < n; i += stride) p[i] = 0.0f;
}

// ---------------------------------------------------------------------------
// Edge-count per (rel, dst) segment (edge structure is layer-invariant)
// ---------------------------------------------------------------------------
__global__ void rgcn_count_kernel(const int* __restrict__ edge_index,
                                  const int* __restrict__ edge_type,
                                  float* __restrict__ cnt,
                                  int n_edges, int n_nodes) {
    long long e = (long long)blockIdx.x * blockDim.x + threadIdx.x;
    if (e >= n_edges) return;
    int d = edge_index[(long long)n_edges + e];
    int r = edge_type[e];
    atomicAdd(cnt + (long long)r * n_nodes + d, 1.0f);
}

__global__ void rgcn_invnorm_kernel(float* __restrict__ cnt, long long n) {
    long long i = (long long)blockIdx.x * blockDim.x + threadIdx.x;
    if (i >= n) return;
    cnt[i] = 1.0f / fmaxf(cnt[i], 1.0f);
}

// ---------------------------------------------------------------------------
// W_r = sum_b comb[r,b] * bases[b]  (r<8);  W_8 = root.  Wall: [9][64][64] f32
// ---------------------------------------------------------------------------
__global__ void rgcn_makew_kernel(const float* __restrict__ bases,
                                  const float* __restrict__ comb,
                                  const float* __restrict__ root,
                                  float* __restrict__ wall) {
    int i = blockIdx.x * blockDim.x + threadIdx.x;
    if (i >= 9 * RGCN_C * RGCN_C) return;
    int r = i >> 12;
    int k = i & 4095;
    float v;
    if (r < RGCN_R) {
        v = 0.0f;
        #pragma unroll
        for (int b = 0; b < RGCN_B; ++b)
            v += comb[r * RGCN_B + b] * bases[b * 4096 + k];
    } else {
        v = root[k];
    }
    wall[i] = v;
}

// ---------------------------------------------------------------------------
// Pack Wall into the v_wmma_f32_16x16x32_bf16 B-operand lane layout:
//   wpack[r][ntile(4)][kstep(2)][lane(32)][j(16)]  (bf16)
// element K = kstep*32 + (lane>=16 ? 8 : 0) + (j<8 ? j : j+8), col = ntile*16 + lane%16
// Each GEMM lane loads one contiguous, 32B-aligned 32-byte fragment.
// ---------------------------------------------------------------------------
__global__ void rgcn_packw_kernel(const float* __restrict__ wall,
                                  bf16_t* __restrict__ wpack) {
    int i = blockIdx.x * blockDim.x + threadIdx.x;
    if (i >= 9 * 4 * 2 * 32 * 16) return;
    int j    = i & 15;
    int lane = (i >> 4) & 31;
    int ks   = (i >> 9) & 1;
    int nt   = (i >> 10) & 3;
    int r    = i >> 12;
    int col  = nt * 16 + (lane & 15);
    int K    = ks * 32 + ((lane >> 4) ? 8 : 0) + (j < 8 ? j : j + 8);
    wpack[i] = f2bf(wall[r * 4096 + K * RGCN_C + col]);
}

// ---------------------------------------------------------------------------
// Edge scatter with norm folded in:
//   h[(r*N+dst)*64 + c] += norm[r*N+dst] * x[src*64 + c]   (packed bf16 atomics)
// 8 threads per edge, 8 channels each: 2x float4 gather -> 4 pk-bf16 atomics.
// ---------------------------------------------------------------------------
__global__ __launch_bounds__(256)
void rgcn_scatter_kernel(const float* __restrict__ xin,
                         const int* __restrict__ edge_index,
                         const int* __restrict__ edge_type,
                         const float* __restrict__ normv,
                         bf16_t* __restrict__ h,
                         int n_edges, int n_nodes) {
    long long idx = (long long)blockIdx.x * blockDim.x + threadIdx.x;
    long long e = idx >> 3;
    int t = (int)(idx & 7);
    if (e >= n_edges) return;
    int s = edge_index[e];
    int d = edge_index[(long long)n_edges + e];
    int r = edge_type[e];
    long long seg = (long long)r * n_nodes + d;
    float nrm = normv[seg];

    const float* sp = xin + (long long)s * RGCN_C + t * 8;
    const float4 f0 = *(const float4*)(sp);
    const float4 f1 = *(const float4*)(sp + 4);

    unsigned p0 = pack2bf(f0.x * nrm, f0.y * nrm);
    unsigned p1 = pack2bf(f0.z * nrm, f0.w * nrm);
    unsigned p2 = pack2bf(f1.x * nrm, f1.y * nrm);
    unsigned p3 = pack2bf(f1.z * nrm, f1.w * nrm);

    bf16_t* dp = h + seg * RGCN_C + t * 8;
    atomic_pk_add_bf16(dp + 0, p0);
    atomic_pk_add_bf16(dp + 2, p1);
    atomic_pk_add_bf16(dp + 4, p2);
    atomic_pk_add_bf16(dp + 6, p3);
}

// ---------------------------------------------------------------------------
// Fused GEMM + bias + ELU.
// Block = 128 threads = 4 waves; wave w owns output columns [16w, 16w+16).
// Tile = 16 nodes per block. 18 WMMAs (8 bf16 relations x 2 K-steps + root x 2).
// A-operand 16-bit 16x32 layout: lanes 0-15 row M=lane, K {0..7,16..23};
// lanes 16-31 row M=lane-16, K {8..15,24..31}.
// ---------------------------------------------------------------------------
__global__ __launch_bounds__(128)
void rgcn_gemm_kernel(const bf16_t* __restrict__ h,     // [R*N*64] bf16 (pre-normed)
                      const float* __restrict__ xin,    // [N*64] f32
                      const bf16_t* __restrict__ wpack, // [9][4][2][32][16]
                      const float* __restrict__ bias,   // [64]
                      float* __restrict__ out,          // [N*64]
                      int n_nodes) {
    const int lane = threadIdx.x & 31;
    const int wave = threadIdx.x >> 5;       // output column tile 0..3
    const int tile = blockIdx.x;
    const int mrow = lane & 15;
    const int hi   = lane >> 4;              // K-half select
    const int node = tile * 16 + mrow;

    v8f acc = {};

    // Relations 0..7: pure bf16 load -> WMMA
    for (int r = 0; r < RGCN_R; ++r) {
        const bf16_t* arow = h + ((long long)r * n_nodes + node) * RGCN_C;
        #pragma unroll
        for (int ks = 0; ks < 2; ++ks) {
            const int klo = ks * 32 + hi * 8;
            const v8bf alo = *(const v8bf*)(arow + klo);
            const v8bf ahi = *(const v8bf*)(arow + klo + 16);
            const v16bf a = __builtin_shufflevector(alo, ahi,
                0, 1, 2, 3, 4, 5, 6, 7, 8, 9, 10, 11, 12, 13, 14, 15);
            const v16bf b = *(const v16bf*)(
                wpack + ((((long long)r * 4 + wave) * 2 + ks) * 32 + lane) * 16);
            acc = __builtin_amdgcn_wmma_f32_16x16x32_bf16(
                false, a, false, b, (short)0, acc, false, false);
        }
    }

    // Root path (relation 8): f32 features, convert in-register
    {
        const float* arow = xin + (long long)node * RGCN_C;
        #pragma unroll
        for (int ks = 0; ks < 2; ++ks) {
            const int klo = ks * 32 + hi * 8;
            const float4 f0 = *(const float4*)(arow + klo);
            const float4 f1 = *(const float4*)(arow + klo + 4);
            const float4 f2 = *(const float4*)(arow + klo + 16);
            const float4 f3 = *(const float4*)(arow + klo + 20);
            v16bf a;
            a[0]  = f2bf(f0.x);  a[1]  = f2bf(f0.y);
            a[2]  = f2bf(f0.z);  a[3]  = f2bf(f0.w);
            a[4]  = f2bf(f1.x);  a[5]  = f2bf(f1.y);
            a[6]  = f2bf(f1.z);  a[7]  = f2bf(f1.w);
            a[8]  = f2bf(f2.x);  a[9]  = f2bf(f2.y);
            a[10] = f2bf(f2.z);  a[11] = f2bf(f2.w);
            a[12] = f2bf(f3.x);  a[13] = f2bf(f3.y);
            a[14] = f2bf(f3.z);  a[15] = f2bf(f3.w);
            const v16bf b = *(const v16bf*)(
                wpack + ((((long long)8 * 4 + wave) * 2 + ks) * 32 + lane) * 16);
            acc = __builtin_amdgcn_wmma_f32_16x16x32_bf16(
                false, a, false, b, (short)0, acc, false, false);
        }
    }

    // C/D layout: VGPR v, lanes 0-15 -> row M=v, lanes 16-31 -> row M=v+8; col = lane%16
    const int col = wave * 16 + mrow;
    const float bv = bias[col];
    #pragma unroll
    for (int v = 0; v < 8; ++v) {
        const int row = tile * 16 + v + 8 * hi;
        float val = acc[v] + bv;
        val = (val > 0.0f) ? val : (__expf(val) - 1.0f);
        out[(long long)row * RGCN_C + col] = val;
    }
}

// ---------------------------------------------------------------------------
// Host-side launch
// ---------------------------------------------------------------------------
extern "C" void kernel_launch(void* const* d_in, const int* in_sizes, int n_in,
                              void* d_out, int out_size, void* d_ws, size_t ws_size,
                              hipStream_t stream) {
    (void)n_in; (void)out_size; (void)ws_size;

    const float* x          = (const float*)d_in[0];
    const int*   edge_index = (const int*)d_in[1];
    const int*   edge_type  = (const int*)d_in[2];
    const int E = in_sizes[2];
    const int N = in_sizes[0] / RGCN_C;

    // Workspace layout (256B-aligned offsets)
    char* ws = (char*)d_ws;
    float*  normv = (float*)(ws + 0);                 // R*N f32           (3.2 MB)
    float*  wall  = (float*)(ws + 3200000);           // 9*4096 f32        (147 KB)
    bf16_t* wpack = (bf16_t*)(ws + 3347456);          // 36864 bf16        (72 KB)
    float*  featA = (float*)(ws + 3421184);           // N*64 f32          (25.6 MB)
    float*  featB = (float*)(ws + 29021184);          // N*64 f32          (25.6 MB)
    bf16_t* hbuf  = (bf16_t*)(ws + 54621184);         // R*N*64 bf16       (102.4 MB)

    const long long RN = (long long)RGCN_R * N;
    const long long H_FLOATS = RN * RGCN_C / 2;       // bf16 buffer in f32 units

    // --- mean-normalization factors (computed once; layer-invariant) ---
    rgcn_zero_kernel<<<2048, 256, 0, stream>>>(normv, RN);
    rgcn_count_kernel<<<(E + 255) / 256, 256, 0, stream>>>(edge_index, edge_type,
                                                           normv, E, N);
    rgcn_invnorm_kernel<<<(unsigned)((RN + 255) / 256), 256, 0, stream>>>(normv, RN);

    const float* cur = x;
    float* outs[4] = { featA, featB, featA, (float*)d_out };

    for (int l = 0; l < 4; ++l) {
        const float* bases = (const float*)d_in[3 + 4 * l];
        const float* comb  = (const float*)d_in[4 + 4 * l];
        const float* root  = (const float*)d_in[5 + 4 * l];
        const float* bias  = (const float*)d_in[6 + 4 * l];

        rgcn_makew_kernel<<<144, 256, 0, stream>>>(bases, comb, root, wall);
        rgcn_packw_kernel<<<144, 256, 0, stream>>>(wall, wpack);

        rgcn_zero_kernel<<<4096, 256, 0, stream>>>((float*)hbuf, H_FLOATS);
        rgcn_scatter_kernel<<<(unsigned)(((long long)E * 8 + 255) / 256), 256, 0, stream>>>(
            cur, edge_index, edge_type, normv, hbuf, E, N);

        rgcn_gemm_kernel<<<N / 16, 128, 0, stream>>>(hbuf, cur, wpack,
                                                     bias, outs[l], N);
        cur = outs[l];
    }
}
